// BidirectionalLSTM_20169166422592
// MI455X (gfx1250) — compile-verified, matrix operand order checked by reading, not run on hardware
//
#include <hip/hip_runtime.h>
#include <hip/hip_bf16.h>

// ---------------------------------------------------------------------------
// Bidirectional LSTM for MI455X (gfx1250, wave32)
//   WMMA bf16 + async-to-LDS (GEMMs) + Tensor Data Mover (recurrent scan)
//   B=64, T=512, NIN=512, H=512, NOUT=512
// ---------------------------------------------------------------------------

typedef __attribute__((ext_vector_type(16))) __bf16 v16bf;
typedef __attribute__((ext_vector_type(8)))  __bf16 v8bf;
typedef __attribute__((ext_vector_type(8)))  float  v8f;
typedef __attribute__((ext_vector_type(4)))  unsigned v4u;
typedef __attribute__((ext_vector_type(8)))  unsigned v8u;

#define WMMA_BF16(a, b, c) \
  __builtin_amdgcn_wmma_f32_16x16x32_bf16(false, (a), false, (b), (short)0, (c), false, false)

// LDS row pitch for staged bf16 slabs: 40 elems = 80B. 16B aligned for b128,
// and gcd(20 dwords, 64 banks) = 4 -> 16 consecutive rows hit 16 distinct banks.
#define SLAB_PITCH 40

// s_wait_asynccnt with a literal immediate
#define WAIT_ASYNCCNT(n) asm volatile("s_wait_asynccnt " #n ::: "memory")

// One 16-byte async copy global -> LDS (ASYNCcnt-tracked, per-lane).
// LDS byte offset = low 32 bits of the generic pointer (aperture mapping).
__device__ __forceinline__ void async_copy_b128(void* lds_dst, const void* gsrc) {
  unsigned lds_off = (unsigned)(uintptr_t)lds_dst;
  asm volatile("global_load_async_to_lds_b128 %0, %1, off"
               :: "v"(lds_off), "v"(gsrc)
               : "memory");
}

// ---------------------------------------------------------------------------
// Tensor Data Mover: one 2-D tile (tile_d0 elems/row x tile_d1 rows, bf16,
// row stride stride0 elems) global -> LDS, with HW padding of 16B per 64B row
// so the LDS image lands at SLAB_PITCH (80B) pitch. TENSORcnt-tracked,
// wave-level (EXEC ignored): call from one wave only.
// D# packing per CDNA5 ISA 8.3/8.4 (group0 128b, group1 256b; VADDR2/3 null).
// ---------------------------------------------------------------------------
__device__ __forceinline__ void tdm_load_2d_padded(unsigned lds_addr,
                                                   unsigned long long gaddr,
                                                   unsigned tensor_d0, unsigned tensor_d1,
                                                   unsigned tile_d0,   unsigned tile_d1,
                                                   unsigned stride0) {
  v4u g0;
  g0[0] = 1u;                                       // count=1 (valid descriptor)
  g0[1] = lds_addr;                                 // lds_addr [63:32]
  g0[2] = (unsigned)gaddr;                          // global_addr [95:64]
  g0[3] = (unsigned)(gaddr >> 32) | (2u << 30);     // global_addr[56:32] | type=2
  v8u g1;
  // workgroup_mask=0 | data_size=1 (2B) | pad_enable | pad_interval=3 (16 dw)
  // | pad_amount=3 (4 dw)
  g1[0] = (1u << 16) | (1u << 20) | (3u << 22) | (3u << 25);   // 0x06D10000
  g1[1] = (tensor_d0 & 0xFFFFu) << 16;              // tensor_dim0[15:0] @ bit48
  g1[2] = ((tensor_d0 >> 16) & 0xFFFFu) | ((tensor_d1 & 0xFFFFu) << 16);
  g1[3] = ((tensor_d1 >> 16) & 0xFFFFu) | (tile_d0 << 16);     // tile_dim0 @ bit112
  g1[4] = tile_d1 & 0xFFFFu;                        // tile_dim1 (tile_dim2=0)
  g1[5] = stride0;                                  // tensor_dim0_stride[31:0]
  g1[6] = 0u;                                       // stride0[47:32]=0, dim1_stride lo
  g1[7] = 0u;
  asm volatile("tensor_load_to_lds %0, %1, null, null"
               :: "s"(g0), "s"(g1)
               : "memory");
}

// Load one 16x32 bf16 operand tile from GLOBAL into the ISA 16-bit A/B layout:
// lane L (half=L/16, idx=L%16) holds row (r0+idx), k-chunks [k0+half*8..+7]
// and [k0+16+half*8..+7]  (two global_load_b128).
__device__ __forceinline__ v16bf load_tile_bf16(const __bf16* __restrict__ base,
                                                long ld, int r0, int k0, int lane) {
  const int half = lane >> 4;
  const int idx  = lane & 15;
  const __bf16* p = base + (size_t)(r0 + idx) * (size_t)ld + (size_t)(k0 + half * 8);
  union { v16bf v; v8bf h[2]; } u;
  u.h[0] = *(const v8bf*)(p);
  u.h[1] = *(const v8bf*)(p + 16);
  return u.v;
}

// Same, but from an LDS slab with SLAB_PITCH row pitch (two ds_load_b128).
__device__ __forceinline__ v16bf lds_load_tile(const __bf16* slab, int r0, int lane) {
  const int half = lane >> 4;
  const int idx  = lane & 15;
  const __bf16* p = slab + (r0 + idx) * SLAB_PITCH + half * 8;
  union { v16bf v; v8bf h[2]; } u;
  u.h[0] = *(const v8bf*)(p);
  u.h[1] = *(const v8bf*)(p + 16);
  return u.v;
}

__device__ __forceinline__ float fast_sigmoid(float x) {
  return 1.0f / (1.0f + __expf(-x));
}
__device__ __forceinline__ float fast_tanh(float x) {
  return 2.0f / (1.0f + __expf(-2.0f * x)) - 1.0f;
}

// ---------------------------------------------------------------------------
// Elementwise helpers
// ---------------------------------------------------------------------------
__global__ void f32_to_bf16_kernel(const float* __restrict__ in,
                                   __bf16* __restrict__ out, int n) {
  for (int i = blockIdx.x * blockDim.x + threadIdx.x; i < n;
       i += gridDim.x * blockDim.x)
    out[i] = (__bf16)in[i];
}

// Permute w_hh (4H, H) f32 into TDM-friendly bf16 layout:
//   out[cg*128 + gate*32 + r32][k] = in[gate*512 + cg*32 + r32][k]
// so each step-kernel block's 128-row slab is contiguous with row stride H.
__global__ void permute_whh_kernel(const float* __restrict__ in,
                                   __bf16* __restrict__ out, int n /* 4H*H */) {
  const int Hd = 512;
  for (int i = blockIdx.x * blockDim.x + threadIdx.x; i < n;
       i += gridDim.x * blockDim.x) {
    const int R = i / Hd, k = i % Hd;
    const int cg = R >> 7, gate = (R >> 5) & 3, r32 = R & 31;
    out[i] = (__bf16)in[(size_t)(gate * 512 + cg * 32 + r32) * Hd + k];
  }
}

__global__ void bias_sum_kernel(const float* __restrict__ a,
                                const float* __restrict__ b,
                                float* __restrict__ out, int n) {
  int i = blockIdx.x * blockDim.x + threadIdx.x;
  if (i < n) out[i] = a[i] + b[i];
}

__global__ void init_state_kernel(float* __restrict__ c, __bf16* __restrict__ h0,
                                  int nc, int nh) {
  int i = blockIdx.x * blockDim.x + threadIdx.x;
  if (i < nc) c[i] = 0.0f;
  if (i < nh) h0[i] = (__bf16)0.0f;
}

// ---------------------------------------------------------------------------
// Generic bf16 WMMA GEMM:  C[m,n] = sum_k A[m,k] * Bw[n,k] + bias[n]
//   block = 128 threads = 4 waves along N, each wave computes a 64x32 tile.
//   The A slab (64 rows x 32 k), shared by all 4 waves, is staged in LDS via
//   double-buffered global_load_async_to_lds_b128; B is per-wave from global.
//   If tposeT != 0, output row m = b*tposeT + t is written at (t * M/tposeT + b)
// ---------------------------------------------------------------------------
__global__ __launch_bounds__(128) void gemm_bf16_kernel(
    const __bf16* __restrict__ A,  size_t a_dir_stride, int lda,
    const __bf16* __restrict__ Bw, size_t b_dir_stride,
    const float*  __restrict__ bias, size_t bias_dir_stride,
    float* __restrict__ C, size_t c_dir_stride,
    int M, int N, int K, int tposeT) {
  __shared__ __bf16 aslab[2][64 * SLAB_PITCH];   // 2 x 5KB

  const int tid  = threadIdx.x;
  const int lane = tid & 31;
  const int wave = tid >> 5;
  const int dir  = blockIdx.z;
  const int m0   = blockIdx.y * 64;
  const int n0   = blockIdx.x * 128 + wave * 32;

  const __bf16* a = A  + a_dir_stride * (size_t)dir;
  const __bf16* b = Bw + b_dir_stride * (size_t)dir;
  float*        c = C  + c_dir_stride * (size_t)dir;

  // 64 rows x 4 x 16B chunks = 256 chunks; 2 per thread
  auto issue_a_slab = [&](int buf, int k) {
#pragma unroll
    for (int t = 0; t < 2; ++t) {
      const int ch  = tid + 128 * t;
      const int row = ch >> 2;
      const int cho = ch & 3;
      async_copy_b128(&aslab[buf][row * SLAB_PITCH + cho * 8],
                      a + (size_t)(m0 + row) * (size_t)lda + (size_t)(k + cho * 8));
    }
  };

  v8f acc[4][2];
#pragma unroll
  for (int i = 0; i < 4; ++i) { acc[i][0] = (v8f)0.0f; acc[i][1] = (v8f)0.0f; }

  int buf = 0;
  issue_a_slab(0, 0);
  for (int k = 0; k < K; k += 32) {
    if (k + 32 < K) {
      issue_a_slab(buf ^ 1, k + 32);          // prefetch next slab
      __builtin_prefetch(b + (size_t)n0 * (size_t)K + (size_t)(k + 32), 0, 1);
      WAIT_ASYNCCNT(2);                       // current slab done, next in flight
    } else {
      WAIT_ASYNCCNT(0);
    }
    __syncthreads();                          // slab[buf] visible to all waves

    v16bf bt0 = load_tile_bf16(b, K, n0,      k, lane);
    v16bf bt1 = load_tile_bf16(b, K, n0 + 16, k, lane);
#pragma unroll
    for (int i = 0; i < 4; ++i) {
      v16bf at = lds_load_tile(aslab[buf], 16 * i, lane);
      acc[i][0] = WMMA_BF16(at, bt0, acc[i][0]);
      acc[i][1] = WMMA_BF16(at, bt1, acc[i][1]);
    }
    __syncthreads();                          // done reading buf before overwrite
    buf ^= 1;
  }

  const int half = lane >> 4, idx = lane & 15;
  const int Brows = tposeT ? (M / tposeT) : 0;
#pragma unroll
  for (int j = 0; j < 2; ++j) {
    const int col = n0 + j * 16 + idx;
    const float bv = bias[bias_dir_stride * (size_t)dir + (size_t)col];
#pragma unroll
    for (int i = 0; i < 4; ++i) {
#pragma unroll
      for (int r = 0; r < 8; ++r) {
        const int m = m0 + 16 * i + r + 8 * half;
        size_t orow = (size_t)m;
        if (tposeT) orow = (size_t)(m % tposeT) * (size_t)Brows + (size_t)(m / tposeT);
        c[orow * (size_t)N + (size_t)col] = acc[i][j][r] + bv;
      }
    }
  }
}

// ---------------------------------------------------------------------------
// Fused bidirectional LSTM step (one launch per time step, both directions).
//   grid = 32 blocks x 128 threads. block = (dir, colgroup cg); waves = M-tile.
//   All 4 waves of a block consume the SAME 128-row w_hh slab (4 gates x 32
//   cols x 32 k). Wave 0 stages it with ONE Tensor-DMA descriptor per k-step
//   (double-buffered, HW-padded to SLAB_PITCH), synchronized with
//   s_wait_tensorcnt + workgroup barrier. Each wave's 16-row h_prev A-tile is
//   a direct global load. Gate math is done in-register.
// ---------------------------------------------------------------------------
__global__ __launch_bounds__(128) void lstm_step_kernel(
    __bf16* __restrict__ hseq,        // (B, T, 2H) bf16, written at pos
    const __bf16* __restrict__ hzero, // (2, B, H) zeros
    const __bf16* __restrict__ wperm, // (2, 16, 128, H) bf16 permuted w_hh
    const float*  __restrict__ xg,    // (2, T, B, 4H) f32 precomputed input gates
    float* __restrict__ cstate,       // (2, B, H) f32 cell state
    int s) {
  const int T = 512, Bn = 64, H = 512;
  __shared__ __bf16 bslab[2][128 * SLAB_PITCH];  // 2 x 10KB

  const int tid  = threadIdx.x;
  const int lane = tid & 31;
  const int mt   = tid >> 5;                // wave = M-tile (16 batch rows)
  const int dir  = blockIdx.x >> 4;
  const int cg   = blockIdx.x & 15;         // 32 hidden units
  const int pos  = dir ? (T - 1 - s) : s;
  const int ppos = dir ? (pos + 1) : (pos - 1);

  // this block's contiguous 128-row weight panel
  const __bf16* wpanel = wperm + ((size_t)dir * 16 + (size_t)cg) * 128 * H;
  const __bf16* aBase;
  long lda;
  if (s == 0) { aBase = hzero + (size_t)dir * Bn * H;            lda = H; }
  else        { aBase = hseq + (size_t)ppos * 2 * H + dir * H;   lda = (long)T * 2 * H; }

  const int m0 = mt * 16;
  v8f acc[4][2];
#pragma unroll
  for (int g = 0; g < 4; ++g) { acc[g][0] = (v8f)0.0f; acc[g][1] = (v8f)0.0f; }

  int buf = 0;
  if (tid < 32)   // wave 0 drives the TDM (EXEC ignored, wave-level op)
    tdm_load_2d_padded((unsigned)(uintptr_t)&bslab[0][0],
                       (unsigned long long)(uintptr_t)wpanel,
                       H, 128, 32, 128, H);
  for (int k = 0; k < H; k += 32) {
    if (tid < 32) {
      if (k + 32 < H) {
        tdm_load_2d_padded((unsigned)(uintptr_t)&bslab[buf ^ 1][0],
                           (unsigned long long)(uintptr_t)(wpanel + k + 32),
                           H, 128, 32, 128, H);
        __builtin_amdgcn_s_wait_tensorcnt(1);  // in-order: current slab done
      } else {
        __builtin_amdgcn_s_wait_tensorcnt(0);
      }
    }
    __syncthreads();                           // slab[buf] visible to all waves

    v16bf at = load_tile_bf16(aBase, lda, m0, k, lane);
#pragma unroll
    for (int g = 0; g < 4; ++g) {
      v16bf b0 = lds_load_tile(bslab[buf], g * 32,      lane);
      v16bf b1 = lds_load_tile(bslab[buf], g * 32 + 16, lane);
      acc[g][0] = WMMA_BF16(at, b0, acc[g][0]);
      acc[g][1] = WMMA_BF16(at, b1, acc[g][1]);
    }
    __syncthreads();                           // done reading buf before overwrite
    buf ^= 1;
  }

  const int half = lane >> 4, idx = lane & 15;
  const float* xgp = xg + ((size_t)dir * T + (size_t)pos) * (size_t)Bn * 4 * H;
  float* cst = cstate + (size_t)dir * Bn * H;

#pragma unroll
  for (int j = 0; j < 2; ++j) {
    const int u = cg * 32 + j * 16 + idx;  // hidden unit within H
#pragma unroll
    for (int r = 0; r < 8; ++r) {
      const int b = m0 + r + 8 * half;     // batch row
      const size_t xoff = (size_t)b * 4 * H + (size_t)u;
      float iv = acc[0][j][r] + xgp[xoff + 0 * H];
      float fv = acc[1][j][r] + xgp[xoff + 1 * H];
      float gv = acc[2][j][r] + xgp[xoff + 2 * H];
      float ov = acc[3][j][r] + xgp[xoff + 3 * H];
      iv = fast_sigmoid(iv);
      fv = fast_sigmoid(fv);
      gv = fast_tanh(gv);
      ov = fast_sigmoid(ov);
      const size_t cidx = (size_t)b * H + (size_t)u;
      float cn = fv * cst[cidx] + iv * gv;
      float hn = ov * fast_tanh(cn);
      cst[cidx] = cn;
      hseq[((size_t)b * T + (size_t)pos) * 2 * H + (size_t)(dir * H) + (size_t)u] = (__bf16)hn;
    }
  }
}

// ---------------------------------------------------------------------------
// Host-side orchestration
// ---------------------------------------------------------------------------
extern "C" void kernel_launch(void* const* d_in, const int* in_sizes, int n_in,
                              void* d_out, int out_size, void* d_ws, size_t ws_size,
                              hipStream_t stream) {
  (void)in_sizes; (void)n_in; (void)out_size; (void)ws_size;
  const int Bn = 64, T = 512, NIN = 512, H = 512, NOUT = 512;
  const int G4 = 4 * H;  // 2048

  const float* x      = (const float*)d_in[0];
  const float* w_ih_f = (const float*)d_in[1];
  const float* w_hh_f = (const float*)d_in[2];
  const float* b_ih_f = (const float*)d_in[3];
  const float* b_hh_f = (const float*)d_in[4];
  const float* w_ih_b = (const float*)d_in[5];
  const float* w_hh_b = (const float*)d_in[6];
  const float* b_ih_b = (const float*)d_in[7];
  const float* b_hh_b = (const float*)d_in[8];
  const float* w_emb  = (const float*)d_in[9];
  const float* b_emb  = (const float*)d_in[10];
  float* out = (float*)d_out;

  // workspace carve-up (all chunks are 256B-aligned sizes)
  char* ws = (char*)d_ws;
  size_t off = 0;
  auto carve = [&](size_t bytes) { char* p = ws + off; off += (bytes + 255) & ~(size_t)255; return p; };
  __bf16* x_bf    = (__bf16*)carve((size_t)Bn * T * NIN * 2);        //  32 MB
  __bf16* wih_bf  = (__bf16*)carve((size_t)2 * G4 * NIN * 2);        //   4 MB
  __bf16* wperm   = (__bf16*)carve((size_t)2 * G4 * H * 2);          //   4 MB (permuted w_hh)
  __bf16* wemb_bf = (__bf16*)carve((size_t)NOUT * 2 * H * 2);        //   1 MB
  float*  bsum    = (float*)carve((size_t)2 * G4 * 4);               //  16 KB
  float*  xg      = (float*)carve((size_t)2 * T * Bn * G4 * 4);      // 512 MB
  __bf16* hseq    = (__bf16*)carve((size_t)Bn * T * 2 * H * 2);      //  64 MB
  float*  cst     = (float*)carve((size_t)2 * Bn * H * 4);           // 256 KB
  __bf16* hzero   = (__bf16*)carve((size_t)2 * Bn * H * 2);          // 128 KB

  // --- 0. zero cell state + zero-h buffer (re-done every call: deterministic)
  init_state_kernel<<<256, 256, 0, stream>>>(cst, hzero, 2 * Bn * H, 2 * Bn * H);

  // --- 1. fp32 -> bf16 conversions (+ w_hh permutation for the TDM panels)
  auto cvt = [&](const float* src, __bf16* dst, int n) {
    int blocks = (n + 255) / 256; if (blocks > 4096) blocks = 4096;
    f32_to_bf16_kernel<<<blocks, 256, 0, stream>>>(src, dst, n);
  };
  cvt(x,      x_bf,                      Bn * T * NIN);
  cvt(w_ih_f, wih_bf,                    G4 * NIN);
  cvt(w_ih_b, wih_bf + (size_t)G4 * NIN, G4 * NIN);
  cvt(w_emb,  wemb_bf,                   NOUT * 2 * H);
  permute_whh_kernel<<<4096, 256, 0, stream>>>(w_hh_f, wperm, G4 * H);
  permute_whh_kernel<<<4096, 256, 0, stream>>>(w_hh_b, wperm + (size_t)G4 * H, G4 * H);

  bias_sum_kernel<<<(G4 + 255) / 256, 256, 0, stream>>>(b_ih_f, b_hh_f, bsum, G4);
  bias_sum_kernel<<<(G4 + 255) / 256, 256, 0, stream>>>(b_ih_b, b_hh_b, bsum + G4, G4);

  // --- 2. input projection: xg[dir, t, b, :] = x[b,t,:] @ w_ih[dir]^T + bsum[dir]
  {
    const int M = Bn * T, N = G4, K = NIN;
    dim3 grid(N / 128, M / 64, 2);
    gemm_bf16_kernel<<<grid, 128, 0, stream>>>(
        x_bf, 0, NIN,
        wih_bf, (size_t)G4 * NIN,
        bsum, (size_t)G4,
        xg, (size_t)T * Bn * G4,
        M, N, K, /*tposeT=*/T);
  }

  // --- 3. recurrent scan: 512 fused bidirectional steps (TDM-staged weights)
  for (int s = 0; s < T; ++s)
    lstm_step_kernel<<<32, 128, 0, stream>>>(hseq, hzero, wperm, xg, cst, s);

  // --- 4. output projection: out = concat(hf,hb) @ w_emb^T + b_emb
  {
    const int M = Bn * T, N = NOUT, K = 2 * H;
    dim3 grid(N / 128, M / 64, 1);
    gemm_bf16_kernel<<<grid, 128, 0, stream>>>(
        hseq, 0, 2 * H,
        wemb_bf, 0,
        b_emb, 0,
        out, 0,
        M, N, K, /*tposeT=*/0);
  }
}
